// LRU_24945170055371
// MI455X (gfx1250) — compile-verified, matrix operand order checked by reading, not run on hardware
//
#include <hip/hip_runtime.h>
#include <hip/hip_bf16.h>
#include <cstdint>

// ---------------------------------------------------------------------------
// LRU layer on gfx1250: Bu = X @ [Bre;Bim]^T (bf16 WMMA), chunked scan
// h = lambda*h + Bu, out = Re(hs C^T) + X*D_skip (bf16 WMMA, fused epilogue).
// BATCH=4, SEQ=4096 -> M=16384 rows; DMODEL=DHID=1024.
// ---------------------------------------------------------------------------

typedef __attribute__((ext_vector_type(16))) __bf16 v16bf;
typedef __attribute__((ext_vector_type(8)))  float  v8f;

#define MT      16384      // BATCH*SEQ
#define DM      1024       // DMODEL
#define HH      1024       // DHID
#define NB      4          // BATCH
#define TT      4096       // SEQ
#define NC      8          // scan chunks
#define TC      512        // SEQ / NC  (= 2^9)

__device__ __forceinline__ unsigned short f2bf(float f) {
    unsigned int u = __float_as_uint(f);
    u += 0x7FFFu + ((u >> 16) & 1u);          // round-to-nearest-even
    return (unsigned short)(u >> 16);
}

union FragU { uint4 q[2]; v16bf v; };

// ---------------------------------------------------------------------------
// WMMA bf16 GEMM:  Out[M,N] (f32) = A[M,K] (bf16 row-major, lda)
//                                 x W[N,K] (bf16 row-major, ldw)
// Fragment layout (ISA 7.12.2, 16-bit, B mirrors A with N<->M):
//   lane&15 -> row (M for A / N for B), half = lane>>4,
//   VGPR0..3: K = 8*half + 0..7 ; VGPR4..7: K = 16 + 8*half + 0..7.
// => per lane: two K-contiguous 16B chunks at +0 and +32B  -> b128 pair.
// Block: 256 thr = 8 waves (2x4); wave tile 32x64; block tile 64x256.
// All per-lane pointers hoisted out of the k-loop; bump by 64B per k-step.
// ---------------------------------------------------------------------------
template <int EPI>
__global__ __launch_bounds__(256)
void lru_gemm_bf16(const unsigned short* __restrict__ A, int lda,
                   const unsigned short* __restrict__ W, int ldw,
                   float* __restrict__ Out, int ldo, int K,
                   const float* __restrict__ skipX,
                   const float* __restrict__ skipD) {
    const int lane  = threadIdx.x & 31;
    const int wave  = threadIdx.x >> 5;
    const int wm    = wave >> 2;                // 0..1
    const int wn    = wave & 3;                 // 0..3
    const int m_base = blockIdx.y * 64  + wm * 32;
    const int n_base = blockIdx.x * 256 + wn * 64;
    const int rlane = lane & 15;
    const int half8 = (lane >> 4) * 8;

    const unsigned short* __restrict__ pa0 = A + (size_t)(m_base + rlane) * lda + half8;
    const unsigned short* __restrict__ pa1 = pa0 + (size_t)16 * lda;
    const unsigned short* __restrict__ pb0 = W + (size_t)(n_base + rlane) * ldw + half8;
    const unsigned short* __restrict__ pb1 = pb0 + (size_t)16 * ldw;
    const unsigned short* __restrict__ pb2 = pb0 + (size_t)32 * ldw;
    const unsigned short* __restrict__ pb3 = pb0 + (size_t)48 * ldw;

    v8f acc[2][4] = {};

#pragma unroll 4
    for (int k0 = 0; k0 < K; k0 += 32) {
        FragU a0, a1, b0, b1, b2, b3;
        a0.q[0] = *reinterpret_cast<const uint4*>(pa0);
        a0.q[1] = *reinterpret_cast<const uint4*>(pa0 + 16);
        a1.q[0] = *reinterpret_cast<const uint4*>(pa1);
        a1.q[1] = *reinterpret_cast<const uint4*>(pa1 + 16);
        b0.q[0] = *reinterpret_cast<const uint4*>(pb0);
        b0.q[1] = *reinterpret_cast<const uint4*>(pb0 + 16);
        b1.q[0] = *reinterpret_cast<const uint4*>(pb1);
        b1.q[1] = *reinterpret_cast<const uint4*>(pb1 + 16);
        b2.q[0] = *reinterpret_cast<const uint4*>(pb2);
        b2.q[1] = *reinterpret_cast<const uint4*>(pb2 + 16);
        b3.q[0] = *reinterpret_cast<const uint4*>(pb3);
        b3.q[1] = *reinterpret_cast<const uint4*>(pb3 + 16);
        pa0 += 32; pa1 += 32; pb0 += 32; pb1 += 32; pb2 += 32; pb3 += 32;

        acc[0][0] = __builtin_amdgcn_wmma_f32_16x16x32_bf16(false, a0.v, false, b0.v, (short)0, acc[0][0], false, false);
        acc[1][0] = __builtin_amdgcn_wmma_f32_16x16x32_bf16(false, a1.v, false, b0.v, (short)0, acc[1][0], false, false);
        acc[0][1] = __builtin_amdgcn_wmma_f32_16x16x32_bf16(false, a0.v, false, b1.v, (short)0, acc[0][1], false, false);
        acc[1][1] = __builtin_amdgcn_wmma_f32_16x16x32_bf16(false, a1.v, false, b1.v, (short)0, acc[1][1], false, false);
        acc[0][2] = __builtin_amdgcn_wmma_f32_16x16x32_bf16(false, a0.v, false, b2.v, (short)0, acc[0][2], false, false);
        acc[1][2] = __builtin_amdgcn_wmma_f32_16x16x32_bf16(false, a1.v, false, b2.v, (short)0, acc[1][2], false, false);
        acc[0][3] = __builtin_amdgcn_wmma_f32_16x16x32_bf16(false, a0.v, false, b3.v, (short)0, acc[0][3], false, false);
        acc[1][3] = __builtin_amdgcn_wmma_f32_16x16x32_bf16(false, a1.v, false, b3.v, (short)0, acc[1][3], false, false);
    }

    // C/D layout: VGPR r, lanes 0-15 -> M=r, lanes 16-31 -> M=8+r; N = lane&15.
    const int half = lane >> 4;
#pragma unroll
    for (int i = 0; i < 2; ++i) {
#pragma unroll
        for (int j = 0; j < 4; ++j) {
#pragma unroll
            for (int r = 0; r < 8; ++r) {
                const int row = m_base + 16 * i + r + 8 * half;
                const int col = n_base + 16 * j + rlane;
                float v = acc[i][j][r];
                if (EPI)
                    v += skipX[(size_t)row * DM + col] * skipD[col];
                Out[(size_t)row * ldo + col] = v;
            }
        }
    }
}

// ---------------------------------------------------------------------------
// Per-h parameters.
// ---------------------------------------------------------------------------
__global__ void lru_params(const float* __restrict__ nu_log,
                           const float* __restrict__ theta_log,
                           float* __restrict__ lam,     // [2H] interleaved re,im
                           float* __restrict__ gamma) { // [H]
    int h = blockIdx.x * blockDim.x + threadIdx.x;
    if (h >= HH) return;
    float nu    = __expf(nu_log[h]);
    float theta = __expf(theta_log[h]);
    float mag   = __expf(-nu);
    lam[2 * h]     = mag * __cosf(theta);
    lam[2 * h + 1] = mag * __sinf(theta);
    gamma[h] = sqrtf(1.0f - __expf(-2.0f * nu) + 1e-5f);
}

__global__ void conv_f32_bf16(const float* __restrict__ X,
                              unsigned short* __restrict__ Y, int n4) {
    int i = blockIdx.x * blockDim.x + threadIdx.x;
    if (i >= n4) return;
    float4 v = reinterpret_cast<const float4*>(X)[i];
    ushort4 o;
    o.x = f2bf(v.x); o.y = f2bf(v.y); o.z = f2bf(v.z); o.w = f2bf(v.w);
    reinterpret_cast<ushort4*>(Y)[i] = o;
}

// Wcat[N=2H, K=D]: rows 0..H-1 = gamma[h]*B_re[h,:], rows H..2H-1 = gamma[h]*B_im[h,:]
__global__ void conv_w(const float* __restrict__ Bre, const float* __restrict__ Bim,
                       const float* __restrict__ gamma,
                       unsigned short* __restrict__ Wcat) {
    int idx = blockIdx.x * blockDim.x + threadIdx.x;   // h*DM + d
    if (idx >= HH * DM) return;
    float g = gamma[idx >> 10];
    Wcat[idx]           = f2bf(Bre[idx] * g);
    Wcat[idx + HH * DM] = f2bf(Bim[idx] * g);
}

// Ccat[N=D, K=2H]: Ccat[d, 0:H] = C_re[d,:], Ccat[d, H:2H] = -C_im[d,:]
__global__ void conv_c(const float* __restrict__ Cre, const float* __restrict__ Cim,
                       unsigned short* __restrict__ Ccat) {
    int idx = blockIdx.x * blockDim.x + threadIdx.x;   // d*HH + h
    if (idx >= DM * HH) return;
    int d = idx >> 10, h = idx & 1023;
    Ccat[(size_t)d * (2 * HH) + h]      = f2bf(Cre[idx]);
    Ccat[(size_t)d * (2 * HH) + HH + h] = f2bf(-Cim[idx]);
}

// ---------------------------------------------------------------------------
// Chunked scan (3 passes). Bu is [m, 2H] f32 (re at h, im at H+h);
// hs written [m, 2H] bf16.  Thread index (c,b,h) with h fastest -> coalesced.
// ---------------------------------------------------------------------------
__global__ __launch_bounds__(256)
void lru_scan_pass1(const float* __restrict__ Bu, const float* __restrict__ lam,
                    float2* __restrict__ Sc) {
    int tid = blockIdx.x * blockDim.x + threadIdx.x;   // (c*NB + b)*HH + h
    int h  = tid & (HH - 1);
    int bc = tid >> 10;
    int b  = bc & (NB - 1);
    int c  = bc >> 2;
    const float lr = lam[2 * h], li = lam[2 * h + 1];
    const float* __restrict__ bu = Bu + ((size_t)b * TT + (size_t)c * TC) * (2 * HH) + h;
    float hr = 0.0f, hi = 0.0f;
#pragma unroll 2
    for (int t = 0; t < TC; ++t) {
        const float br = bu[0];
        const float bi = bu[HH];
        const float nr = fmaf(lr, hr, fmaf(-li, hi, br));
        const float ni = fmaf(lr, hi, fmaf(li, hr, bi));
        hr = nr; hi = ni;
        bu += 2 * HH;
    }
    Sc[tid] = make_float2(hr, hi);
}

__global__ void lru_scan_pass2(const float2* __restrict__ Sc,
                               const float* __restrict__ lam,
                               float2* __restrict__ Carr) {
    int tid = blockIdx.x * blockDim.x + threadIdx.x;   // b*HH + h
    if (tid >= NB * HH) return;
    int h = tid & (HH - 1);
    int b = tid >> 10;
    const float lr = lam[2 * h], li = lam[2 * h + 1];
    // lambda^TC via 9 complex squarings (TC = 2^9)
    float wr = lr, wi = li;
#pragma unroll
    for (int i = 0; i < 9; ++i) {
        float nr = wr * wr - wi * wi;
        float ni = 2.0f * wr * wi;
        wr = nr; wi = ni;
    }
    float cr = 0.0f, ci = 0.0f;
#pragma unroll
    for (int c = 0; c < NC; ++c) {
        int idx = (c * NB + b) * HH + h;
        Carr[idx] = make_float2(cr, ci);
        float2 s = Sc[idx];
        float nr = fmaf(wr, cr, fmaf(-wi, ci, s.x));
        float ni = fmaf(wr, ci, fmaf(wi, cr, s.y));
        cr = nr; ci = ni;
    }
}

__global__ __launch_bounds__(256)
void lru_scan_pass3(const float* __restrict__ Bu, const float* __restrict__ lam,
                    const float2* __restrict__ Carr,
                    unsigned short* __restrict__ Hs) {
    int tid = blockIdx.x * blockDim.x + threadIdx.x;   // (c*NB + b)*HH + h
    int h  = tid & (HH - 1);
    int bc = tid >> 10;
    int b  = bc & (NB - 1);
    int c  = bc >> 2;
    const float lr = lam[2 * h], li = lam[2 * h + 1];
    const size_t base = ((size_t)b * TT + (size_t)c * TC) * (2 * HH) + h;
    const float* __restrict__ bu  = Bu + base;
    unsigned short* __restrict__ out = Hs + base;
    float2 c0 = Carr[tid];
    float hr = c0.x, hi = c0.y;
#pragma unroll 2
    for (int t = 0; t < TC; ++t) {
        const float br = bu[0];
        const float bi = bu[HH];
        const float nr = fmaf(lr, hr, fmaf(-li, hi, br));
        const float ni = fmaf(lr, hi, fmaf(li, hr, bi));
        hr = nr; hi = ni;
        out[0]  = f2bf(hr);
        out[HH] = f2bf(hi);
        bu  += 2 * HH;
        out += 2 * HH;
    }
}

// ---------------------------------------------------------------------------
extern "C" void kernel_launch(void* const* d_in, const int* in_sizes, int n_in,
                              void* d_out, int out_size, void* d_ws, size_t ws_size,
                              hipStream_t stream) {
    const float* inputs    = (const float*)d_in[0];   // [4,4096,1024]
    const float* nu_log    = (const float*)d_in[1];
    const float* theta_log = (const float*)d_in[2];
    const float* B_re      = (const float*)d_in[3];
    const float* B_im      = (const float*)d_in[4];
    const float* C_re      = (const float*)d_in[5];
    const float* C_im      = (const float*)d_in[6];
    const float* D_skip    = (const float*)d_in[7];
    float* out = (float*)d_out;

    char* ws = (char*)d_ws;
    size_t off = 0;
    auto carve = [&](size_t bytes) {
        char* p = ws + off;
        off = (off + bytes + 255) & ~(size_t)255;
        return p;
    };
    unsigned short* Xb   = (unsigned short*)carve((size_t)MT * DM * 2);        // 32 MB
    unsigned short* Wcat = (unsigned short*)carve((size_t)2 * HH * DM * 2);    //  4 MB
    unsigned short* Ccat = (unsigned short*)carve((size_t)DM * 2 * HH * 2);    //  4 MB
    float*          lam  = (float*)carve((size_t)2 * HH * 4);
    float*          gam  = (float*)carve((size_t)HH * 4);
    float2*         Sc   = (float2*)carve((size_t)NC * NB * HH * 8);
    float2*         Carr = (float2*)carve((size_t)NC * NB * HH * 8);
    float*          Bu   = (float*)carve((size_t)MT * 2 * HH * 4);             // 128 MB
    unsigned short* Hs   = (unsigned short*)carve((size_t)MT * 2 * HH * 2);    //  64 MB
    (void)ws_size; (void)in_sizes; (void)n_in; (void)out_size;

    // 1) per-h parameters
    lru_params<<<(HH + 255) / 256, 256, 0, stream>>>(nu_log, theta_log, lam, gam);
    // 2) conversions
    conv_f32_bf16<<<(MT * DM / 4 + 255) / 256, 256, 0, stream>>>(inputs, Xb, MT * DM / 4);
    conv_w<<<(HH * DM + 255) / 256, 256, 0, stream>>>(B_re, B_im, gam, Wcat);
    conv_c<<<(DM * HH + 255) / 256, 256, 0, stream>>>(C_re, C_im, Ccat);
    // 3) GEMM1: Bu[M, 2H] = Xb[M, D] x Wcat[2H, D]^T
    lru_gemm_bf16<0><<<dim3(2 * HH / 256, MT / 64), 256, 0, stream>>>(
        Xb, DM, Wcat, DM, Bu, 2 * HH, DM, nullptr, nullptr);
    // 4) chunked recurrence
    lru_scan_pass1<<<(NC * NB * HH) / 256, 256, 0, stream>>>(Bu, lam, Sc);
    lru_scan_pass2<<<(NB * HH + 255) / 256, 256, 0, stream>>>(Sc, lam, Carr);
    lru_scan_pass3<<<(NC * NB * HH) / 256, 256, 0, stream>>>(Bu, lam, Carr, Hs);
    // 5) GEMM2: out[M, D] = Hs[M, 2H] x Ccat[D, 2H]^T + inputs * D_skip
    lru_gemm_bf16<1><<<dim3(DM / 256, MT / 64), 256, 0, stream>>>(
        Hs, 2 * HH, Ccat, 2 * HH, out, DM, 2 * HH, inputs, D_skip);
}
